// SparseMoEBlock_36764920054145
// MI455X (gfx1250) — compile-verified, matrix operand order checked by reading, not run on hardware
//
#include <hip/hip_runtime.h>
#include <hip/hip_bf16.h>
#include <math.h>

// ---------------- problem constants (from reference) ----------------
#define T_TOK 4096   // B*S
#define H_DIM 1024
#define I_DIM 4096
#define E_NUM 8
// TOP_K = 2

// ---------------- workspace layout (bytes) ----------------
// counts[8] | offsets[8] | dense weights [T,E] f32 | idx lists [E,T] i32 | h bf16 [T*K, I]
#define OFF_COUNTS   0
#define OFF_OFFSETS  128
#define OFF_WD       4096                       // 4096*8*4   = 131072
#define OFF_IDX      (4096 + 131072)            // 8*4096*4   = 131072
#define OFF_H        (512 * 1024)               // 8192*4096*2 = 64 MiB
// total ws use: ~67.6 MB

typedef __bf16 bf16_t;
typedef __attribute__((ext_vector_type(16))) __bf16 v16bf;
typedef __attribute__((ext_vector_type(8)))  __bf16 v8bf;
typedef __attribute__((ext_vector_type(2)))  __bf16 v2bf;
typedef __attribute__((ext_vector_type(8)))  float  v8f;
typedef __attribute__((ext_vector_type(4)))  int    v4i;

// ---------------- fp32 -> bf16 (prefer HW packed convert) ----------------
#if defined(__has_builtin)
#if __has_builtin(__builtin_amdgcn_cvt_pk_bf16_f32)
#define HAVE_CVT_PK_BF16 1
#endif
#endif

__device__ __forceinline__ unsigned pk2(float a, float b) {
#ifdef HAVE_CVT_PK_BF16
  v2bf p = __builtin_amdgcn_cvt_pk_bf16_f32(a, b);  // lo = bf16(a), hi = bf16(b)
  union { v2bf v; unsigned u; } c; c.v = p;
  return c.u;
#else
  union { float f; unsigned u; } ua, ub;
  ua.f = a; ub.f = b;
  unsigned ra = ua.u + 0x7FFFu + ((ua.u >> 16) & 1u);
  unsigned rb = ub.u + 0x7FFFu + ((ub.u >> 16) & 1u);
  return (ra >> 16) | (rb & 0xFFFF0000u);
#endif
}
__device__ __forceinline__ unsigned short f2bf(float f) {
  return (unsigned short)(pk2(f, 0.f) & 0xFFFFu);
}

// ---------------- CDNA5 async global->LDS copy (ASYNCcnt path) ----------------
#if defined(__has_builtin)
#if __has_builtin(__builtin_amdgcn_global_load_async_to_lds_b128)
#define HAVE_ASYNC_LDS 1
#endif
#endif

__device__ __forceinline__ void async_cp16(const void* g, void* l) {
#ifdef HAVE_ASYNC_LDS
  __builtin_amdgcn_global_load_async_to_lds_b128(
      (__attribute__((address_space(1))) v4i*)g,
      (__attribute__((address_space(3))) v4i*)l, 0, 0);
#else
  *(uint4*)l = *(const uint4*)g;
#endif
}
__device__ __forceinline__ void async_wait0() {
#ifdef HAVE_ASYNC_LDS
#if __has_builtin(__builtin_amdgcn_s_wait_asynccnt)
  __builtin_amdgcn_s_wait_asynccnt(0);
#else
  asm volatile("s_wait_asynccnt 0" ::: "memory");
#endif
#endif
}

// Load one 16x32 bf16 fragment row-chunk pair from LDS (row-major, stride 40 halves)
__device__ __forceinline__ v16bf ld_frag(const unsigned short* smem, int row, int k0) {
  v8bf lo = *(const v8bf*)(&smem[row * 40 + k0]);
  v8bf hi = *(const v8bf*)(&smem[row * 40 + k0 + 16]);
  v16bf r;
#pragma unroll
  for (int i = 0; i < 8; ++i) { r[i] = lo[i]; r[i + 8] = hi[i]; }
  return r;
}

// ---------------- kernel 1: zero output + counts ----------------
__global__ __launch_bounds__(256) void moe_init(float* __restrict__ out, int* __restrict__ counts) {
  size_t i = ((size_t)blockIdx.x * 256 + threadIdx.x) * 4;
  if (i < (size_t)T_TOK * H_DIM) *(float4*)(out + i) = make_float4(0.f, 0.f, 0.f, 0.f);
  if (blockIdx.x == 0 && threadIdx.x < E_NUM) counts[threadIdx.x] = 0;
}

// ---------------- kernel 2: router (wave per token) ----------------
__global__ __launch_bounds__(256) void moe_router(const float* __restrict__ x,
                                                  const float* __restrict__ Wg,
                                                  int* __restrict__ counts,
                                                  int* __restrict__ idxl,
                                                  float* __restrict__ wd) {
  const int lane = threadIdx.x & 31;
  const int t = blockIdx.x * 8 + (threadIdx.x >> 5);
  const float* xr = x + (size_t)t * H_DIM;
  float a[E_NUM];
#pragma unroll
  for (int e = 0; e < E_NUM; ++e) a[e] = 0.f;
  for (int h = lane; h < H_DIM; h += 32) {
    float xv = xr[h];
#pragma unroll
    for (int e = 0; e < E_NUM; ++e) a[e] += xv * Wg[e * H_DIM + h];
  }
#pragma unroll
  for (int e = 0; e < E_NUM; ++e)
#pragma unroll
    for (int off = 16; off > 0; off >>= 1) a[e] += __shfl_xor(a[e], off, 32);

  if (lane == 0) {
    float mx = a[0];
#pragma unroll
    for (int e = 1; e < E_NUM; ++e) mx = fmaxf(mx, a[e]);
    float p[E_NUM];
#pragma unroll
    for (int e = 0; e < E_NUM; ++e) p[e] = __expf(a[e] - mx);
    int i1 = 0;
#pragma unroll
    for (int e = 1; e < E_NUM; ++e) if (p[e] > p[i1]) i1 = e;
    int i2 = (i1 == 0) ? 1 : 0;
#pragma unroll
    for (int e = 0; e < E_NUM; ++e) if (e != i1 && p[e] > p[i2]) i2 = e;
    float w1 = p[i1], w2 = p[i2];
    float iw = 1.f / (w1 + w2);   // softmax denom cancels in top-k renorm
    w1 *= iw; w2 *= iw;
#pragma unroll
    for (int e = 0; e < E_NUM; ++e) wd[t * E_NUM + e] = 0.f;
    wd[t * E_NUM + i1] = w1;
    wd[t * E_NUM + i2] = w2;
    int p1 = atomicAdd(&counts[i1], 1); idxl[i1 * T_TOK + p1] = t;
    int p2 = atomicAdd(&counts[i2], 1); idxl[i2 * T_TOK + p2] = t;
  }
}

// ---------------- kernel 3: slot offsets (E=8 prefix sum) ----------------
__global__ void moe_offsets(const int* __restrict__ counts, int* __restrict__ offs) {
  if (threadIdx.x == 0) {
    int run = 0;
    for (int e = 0; e < E_NUM; ++e) { offs[e] = run; run += counts[e]; }
  }
}

// ---------------- kernel 4: FFN layer 1: h = gelu(Xe @ W1[e]^T + b1[e]) ----------------
// block tile 64 (tokens) x 128 (I cols), 8 waves, each wave 32x32 (2x2 wmma frags)
__global__ __launch_bounds__(256) void moe_ffn1(const float* __restrict__ x,
                                                const float* __restrict__ W1,
                                                const float* __restrict__ b1,
                                                const int* __restrict__ counts,
                                                const int* __restrict__ offs,
                                                const int* __restrict__ idxl,
                                                unsigned short* __restrict__ hbuf) {
  __shared__ unsigned short As[64 * 40];    // 64 rows x 32 halves, stride 40
  __shared__ unsigned short Bs[128 * 40];
  const int e = blockIdx.z;
  const int mt = blockIdx.y;
  const int nb = blockIdx.x * 128;
  const int cnt = counts[e];
  if (mt * 64 >= cnt) return;
  const int rows = min(64, cnt - mt * 64);

  const int tid = threadIdx.x;
  const int lane = tid & 31;
  const int w = tid >> 5;
  const int wm = w >> 2, wn = w & 3;
  const int k0 = (lane >> 4) * 8;

  // staging roles
  const int ar = tid >> 2, ac = (tid & 3) * 8;   // A: 64 rows x 8 halves
  const int br = tid >> 1, bc = (tid & 1) * 16;  // B: 128 rows x 16 halves
  int tokA = 0;
  if (ar < rows) tokA = idxl[e * T_TOK + mt * 64 + ar];
  const float* xrow = x + (size_t)tokA * H_DIM;
  const float* w1row = W1 + ((size_t)e * I_DIM + nb + br) * H_DIM;

  v8f acc[2][2] = {};

  for (int kk = 0; kk < H_DIM; kk += 32) {
    if (kk + 32 < H_DIM) __builtin_prefetch(w1row + kk + 32 + bc, 0, 3);
    uint4 qa = make_uint4(0u, 0u, 0u, 0u);
    if (ar < rows) {
      float4 f0 = *(const float4*)(xrow + kk + ac);
      float4 f1 = *(const float4*)(xrow + kk + ac + 4);
      qa = make_uint4(pk2(f0.x, f0.y), pk2(f0.z, f0.w), pk2(f1.x, f1.y), pk2(f1.z, f1.w));
    }
    *(uint4*)(&As[ar * 40 + ac]) = qa;
    float4 g0 = *(const float4*)(w1row + kk + bc);
    float4 g1 = *(const float4*)(w1row + kk + bc + 4);
    float4 g2 = *(const float4*)(w1row + kk + bc + 8);
    float4 g3 = *(const float4*)(w1row + kk + bc + 12);
    *(uint4*)(&Bs[br * 40 + bc])     = make_uint4(pk2(g0.x, g0.y), pk2(g0.z, g0.w), pk2(g1.x, g1.y), pk2(g1.z, g1.w));
    *(uint4*)(&Bs[br * 40 + bc + 8]) = make_uint4(pk2(g2.x, g2.y), pk2(g2.z, g2.w), pk2(g3.x, g3.y), pk2(g3.z, g3.w));
    __syncthreads();

#pragma unroll
    for (int fm = 0; fm < 2; ++fm) {
      v16bf afrag = ld_frag(As, wm * 32 + fm * 16 + (lane & 15), k0);
#pragma unroll
      for (int fn = 0; fn < 2; ++fn) {
        v16bf bfrag = ld_frag(Bs, wn * 32 + fn * 16 + (lane & 15), k0);
        acc[fm][fn] = __builtin_amdgcn_wmma_f32_16x16x32_bf16(
            false, afrag, false, bfrag, (short)0, acc[fm][fn], false, false);
      }
    }
    __syncthreads();
  }

  const int slotBase = offs[e] + mt * 64;
#pragma unroll
  for (int fm = 0; fm < 2; ++fm)
#pragma unroll
    for (int fn = 0; fn < 2; ++fn)
#pragma unroll
      for (int v = 0; v < 8; ++v) {
        int m = wm * 32 + fm * 16 + (lane >> 4) * 8 + v;
        int n = wn * 32 + fn * 16 + (lane & 15);
        if (m < rows) {
          float val = acc[fm][fn][v] + b1[e * I_DIM + nb + n];
          val = 0.5f * val * (1.0f + erff(val * 0.70710678118f));   // exact gelu
          hbuf[(size_t)(slotBase + m) * I_DIM + nb + n] = f2bf(val);
        }
      }
}

// ---------------- kernel 5: FFN layer 2 + weighted combine ----------------
// y = h @ W2[e]^T + b2[e];  out[tok] += w(tok,e) * y   (f32 global atomics)
__global__ __launch_bounds__(256) void moe_ffn2(const float* __restrict__ W2,
                                                const float* __restrict__ b2,
                                                const unsigned short* __restrict__ hbuf,
                                                const int* __restrict__ counts,
                                                const int* __restrict__ offs,
                                                const int* __restrict__ idxl,
                                                const float* __restrict__ wd,
                                                float* __restrict__ out) {
  __shared__ unsigned short As[64 * 40];
  __shared__ unsigned short Bs[128 * 40];
  const int e = blockIdx.z;
  const int mt = blockIdx.y;
  const int nb = blockIdx.x * 128;   // over H
  const int cnt = counts[e];
  if (mt * 64 >= cnt) return;
  const int rows = min(64, cnt - mt * 64);

  const int tid = threadIdx.x;
  const int lane = tid & 31;
  const int w = tid >> 5;
  const int wm = w >> 2, wn = w & 3;
  const int k0 = (lane >> 4) * 8;

  const int ar = tid >> 2, ac = (tid & 3) * 8;
  const int br = tid >> 1, bc = (tid & 1) * 16;
  const int arc = (ar < rows) ? ar : (rows - 1);   // clamp for address safety
  const unsigned short* hrow = hbuf + (size_t)(offs[e] + mt * 64 + arc) * I_DIM;
  const float* w2row = W2 + ((size_t)e * H_DIM + nb + br) * I_DIM;

  v8f acc[2][2] = {};

  for (int kk = 0; kk < I_DIM; kk += 32) {
    if (kk + 32 < I_DIM) __builtin_prefetch(w2row + kk + 32 + bc, 0, 3);
    // A is already bf16 -> async DMA straight into LDS (CDNA5 ASYNCcnt path)
    async_cp16(hrow + kk + ac, &As[ar * 40 + ac]);
    // B: fp32 weights -> convert in registers -> LDS
    float4 g0 = *(const float4*)(w2row + kk + bc);
    float4 g1 = *(const float4*)(w2row + kk + bc + 4);
    float4 g2 = *(const float4*)(w2row + kk + bc + 8);
    float4 g3 = *(const float4*)(w2row + kk + bc + 12);
    *(uint4*)(&Bs[br * 40 + bc])     = make_uint4(pk2(g0.x, g0.y), pk2(g0.z, g0.w), pk2(g1.x, g1.y), pk2(g1.z, g1.w));
    *(uint4*)(&Bs[br * 40 + bc + 8]) = make_uint4(pk2(g2.x, g2.y), pk2(g2.z, g2.w), pk2(g3.x, g3.y), pk2(g3.z, g3.w));
    async_wait0();          // our wave's async copies landed in LDS
    __syncthreads();        // everyone's copies visible

#pragma unroll
    for (int fm = 0; fm < 2; ++fm) {
      v16bf afrag = ld_frag(As, wm * 32 + fm * 16 + (lane & 15), k0);
#pragma unroll
      for (int fn = 0; fn < 2; ++fn) {
        v16bf bfrag = ld_frag(Bs, wn * 32 + fn * 16 + (lane & 15), k0);
        acc[fm][fn] = __builtin_amdgcn_wmma_f32_16x16x32_bf16(
            false, afrag, false, bfrag, (short)0, acc[fm][fn], false, false);
      }
    }
    __syncthreads();
  }

  const int lbase = mt * 64;
#pragma unroll
  for (int fm = 0; fm < 2; ++fm)
#pragma unroll
    for (int fn = 0; fn < 2; ++fn)
#pragma unroll
      for (int v = 0; v < 8; ++v) {
        int m = wm * 32 + fm * 16 + (lane >> 4) * 8 + v;
        int n = wn * 32 + fn * 16 + (lane & 15);
        if (m < rows) {
          int tok = idxl[e * T_TOK + lbase + m];
          float wgt = wd[tok * E_NUM + e];
          float val = (acc[fm][fn][v] + b2[e * H_DIM + nb + n]) * wgt;
          atomicAdd(&out[(size_t)tok * H_DIM + nb + n], val);
        }
      }
}

// ---------------- launcher ----------------
extern "C" void kernel_launch(void* const* d_in, const int* in_sizes, int n_in,
                              void* d_out, int out_size, void* d_ws, size_t ws_size,
                              hipStream_t stream) {
  (void)in_sizes; (void)n_in; (void)out_size; (void)ws_size;
  const float* x  = (const float*)d_in[0];
  const float* Wg = (const float*)d_in[1];
  const float* W1 = (const float*)d_in[2];
  const float* b1 = (const float*)d_in[3];
  const float* W2 = (const float*)d_in[4];
  const float* b2 = (const float*)d_in[5];
  float* out = (float*)d_out;
  char* ws = (char*)d_ws;
  int*            counts = (int*)(ws + OFF_COUNTS);
  int*            offs   = (int*)(ws + OFF_OFFSETS);
  float*          wd     = (float*)(ws + OFF_WD);
  int*            idxl   = (int*)(ws + OFF_IDX);
  unsigned short* hbuf   = (unsigned short*)(ws + OFF_H);

  moe_init<<<dim3((T_TOK * H_DIM) / 1024), dim3(256), 0, stream>>>(out, counts);
  moe_router<<<dim3(T_TOK / 8), dim3(256), 0, stream>>>(x, Wg, counts, idxl, wd);
  moe_offsets<<<dim3(1), dim3(32), 0, stream>>>(counts, offs);
  moe_ffn1<<<dim3(I_DIM / 128, T_TOK / 64, E_NUM), dim3(256), 0, stream>>>(
      x, W1, b1, counts, offs, idxl, hbuf);
  moe_ffn2<<<dim3(H_DIM / 128, T_TOK / 64, E_NUM), dim3(256), 0, stream>>>(
      W2, b2, hbuf, counts, offs, idxl, wd, out);
}
